// RoPEAttention_37117107372119
// MI455X (gfx1250) — matrix operand order repaired and checked by reading, not compiled, not run on hardware
//
#include <hip/hip_runtime.h>

typedef _Float16 h16;
typedef __attribute__((ext_vector_type(16))) _Float16 v16h;
typedef __attribute__((ext_vector_type(8)))  _Float16 v8h;
typedef __attribute__((ext_vector_type(4)))  _Float16 v4h;
typedef __attribute__((ext_vector_type(8)))  float    v8f;
typedef __attribute__((ext_vector_type(4)))  float    v4f;

#define WMMA_F16(A,B,C) __builtin_amdgcn_wmma_f32_16x16x32_f16(false,(A),false,(B),(short)0,(C),false,false)

// Build a v16h A/B fragment from two contiguous 8-half (128-bit) chunks.
__device__ __forceinline__ v16h ld_pair(const h16* p0, const h16* p1) {
  v8h a = *(const v8h*)p0;
  v8h b = *(const v8h*)p1;
  v16h r;
#pragma unroll
  for (int j = 0; j < 8; ++j) { r[j] = a[j]; r[j + 8] = b[j]; }
  return r;
}

// ---------------------------------------------------------------------------
// Kernel 1: qkv = x @ W_qkv, fused RoPE on q/k, q pre-scaled by 1/sqrt(HD).
// Output layout: Q/K/V f16 [b][h][n][64].
// Grid (512, 9), 128 threads (4 waves). Block: 32 rows x 256 cols.
// Each wave: 32 rows x 64 cols (one head), 8 WMMA per K-step.
// ---------------------------------------------------------------------------
__global__ __launch_bounds__(128) void qkv_rope_kernel(
    const float* __restrict__ x, const float* __restrict__ sinp,
    const float* __restrict__ cosp, const float* __restrict__ Wqkv,
    h16* __restrict__ Qws, h16* __restrict__ Kws, h16* __restrict__ Vws) {
  __shared__ __align__(16) h16 ldsA[32][32];     // x tile (f16)
  __shared__ __align__(16) h16 ldsBt[256][40];   // W tile, transposed, padded
  const int tid  = threadIdx.x;
  const int lane = tid & 31, w = tid >> 5;
  const int ln = lane & 15, hi = lane >> 4;
  const int r0  = blockIdx.x * 32;               // row block (b*1024+n)
  const int n0b = blockIdx.y * 256;              // block col base
  const int n0  = n0b + w * 64;                  // wave col base (head aligned)

  v8f acc[2][4] = {};
  for (int k0 = 0; k0 < 768; k0 += 32) {
    __syncthreads();
#pragma unroll
    for (int i = 0; i < 2; ++i) {                // stage A: 32x32, float4 loads
      int idx4 = (tid + i * 128) * 4;
      int m = idx4 >> 5, kk = idx4 & 31;
      v4f xa = *(const v4f*)&x[(size_t)(r0 + m) * 768 + k0 + kk];
      v4h ha;
#pragma unroll
      for (int j = 0; j < 4; ++j) ha[j] = (h16)xa[j];
      *(v4h*)&ldsA[m][kk] = ha;                  // ds_store_b64
    }
    for (int i = 0; i < 16; ++i) {               // stage B^T: 32x256, float4
      int idx4 = (tid + i * 128) * 4;
      int kk = idx4 >> 8, c = idx4 & 255;
      v4f wv = *(const v4f*)&Wqkv[(size_t)(k0 + kk) * 2304 + n0b + c];
#pragma unroll
      for (int j = 0; j < 4; ++j) ldsBt[c + j][kk] = (h16)wv[j];
    }
    if (k0 + 32 < 768)                           // hint next W tile into caches
      __builtin_prefetch(&Wqkv[(size_t)(k0 + 32) * 2304 + n0b + tid * 4], 0, 1);
    __syncthreads();

    // batch all fragment loads, then WMMA back-to-back
    v16h a0 = ld_pair(&ldsA[ln][hi * 8],      &ldsA[ln][hi * 8 + 16]);
    v16h a1 = ld_pair(&ldsA[16 + ln][hi * 8], &ldsA[16 + ln][hi * 8 + 16]);
    v16h bf[4];
#pragma unroll
    for (int t = 0; t < 4; ++t) {
      const h16* bp = &ldsBt[w * 64 + t * 16 + ln][hi * 8];
      bf[t] = ld_pair(bp, bp + 16);
    }
#pragma unroll
    for (int t = 0; t < 4; ++t) acc[0][t] = WMMA_F16(a0, bf[t], acc[0][t]);
#pragma unroll
    for (int t = 0; t < 4; ++t) acc[1][t] = WMMA_F16(a1, bf[t], acc[1][t]);
  }

  // Epilogue: acc[u][t][r] is (row = u*16 + r + 8*hi, col = n0 + 16t + ln).
  const int which = n0 / 768;                    // 0=q 1=k 2=v
  const int hh    = (n0 % 768) / 64;
  const int b     = r0 >> 10;
  const int tok0  = r0 & 1023;
  h16* dst  = (which == 0) ? Qws : (which == 1 ? Kws : Vws);
  h16* base = dst + (size_t)(b * 12 + hh) * 1024 * 64;

#pragma unroll
  for (int u = 0; u < 2; ++u) {
    if (which < 2) {                             // RoPE: partner col is d ^ 32
#pragma unroll
      for (int t = 0; t < 4; ++t) {
        const int d = t * 16 + ln;               // within-head dim
#pragma unroll
        for (int r = 0; r < 8; ++r) {
          const int tok = tok0 + u * 16 + r + hi * 8;
          const float cs = cosp[tok * 64 + d];
          const float sn = sinp[tok * 64 + d];
          const float rot = (t < 2) ? -acc[u][t + 2][r] : acc[u][t - 2][r];
          float v = acc[u][t][r] * cs + rot * sn;
          if (which == 0) v *= 0.125f;           // 1/sqrt(64) folded into Q
          base[(size_t)tok * 64 + d] = (h16)v;
        }
      }
    } else {
#pragma unroll
      for (int t = 0; t < 4; ++t) {
        const int d = t * 16 + ln;
#pragma unroll
        for (int r = 0; r < 8; ++r) {
          const int tok = tok0 + u * 16 + r + hi * 8;
          base[(size_t)tok * 64 + d] = (h16)acc[u][t][r];
        }
      }
    }
  }
}

// ---------------------------------------------------------------------------
// Kernel 2: flash attention. Grid (16, 12, 16), 128 threads.
// Each wave: 16 q rows; loops 32-key tiles with online softmax.
// ---------------------------------------------------------------------------
__global__ __launch_bounds__(128) void attn_kernel(
    const h16* __restrict__ Qws, const h16* __restrict__ Kws,
    const h16* __restrict__ Vws, h16* __restrict__ Ows) {
  __shared__ __align__(16) h16 ldsVt[64][40];     // V tile transposed [d][key]
  __shared__ __align__(16) h16 ldsP[4][16][40];   // per-wave P staging
  const int tid  = threadIdx.x;
  const int lane = tid & 31, w = tid >> 5;
  const int ln = lane & 15, hi = lane >> 4;
  const int b = blockIdx.z, hh = blockIdx.y;
  const int q0 = blockIdx.x * 64 + w * 16;
  const size_t bh = (size_t)(b * 12 + hh) * 1024;
  const h16* Qb = Qws + bh * 64;
  const h16* Kb = Kws + bh * 64;
  const h16* Vb = Vws + bh * 64;

  // Q fragments for contraction chunks d0=0 and d0=32 (reused all tiles).
  const h16* qrow = Qb + (size_t)(q0 + ln) * 64;
  const v16h qa0 = ld_pair(qrow + hi * 16,      qrow + hi * 16 + 8);
  const v16h qa1 = ld_pair(qrow + 32 + hi * 16, qrow + 32 + hi * 16 + 8);

  v8f o[4] = {};
  float mrow[8], lrow[8];
#pragma unroll
  for (int r = 0; r < 8; ++r) { mrow[r] = -1e30f; lrow[r] = 0.0f; }

  for (int key0 = 0; key0 < 1024; key0 += 32) {
    __syncthreads();                              // prev tile's Vt reads done
#pragma unroll
    for (int i = 0; i < 4; ++i) {                 // stage V tile transposed
      int idx4 = (tid + i * 128) * 4;             // v4h global loads
      int kk = idx4 >> 6, d = idx4 & 63;
      v4h vv = *(const v4h*)&Vb[(size_t)(key0 + kk) * 64 + d];
#pragma unroll
      for (int j = 0; j < 4; ++j) ldsVt[d + j][kk] = vv[j];
    }

    // scores S = Q @ K^T for 32 keys: batch 4 K frags then 4 WMMA
    v16h kf[4];
    {
      const h16* kr0 = Kb + (size_t)(key0 + ln) * 64;
      const h16* kr1 = Kb + (size_t)(key0 + 16 + ln) * 64;
      kf[0] = ld_pair(kr0 + hi * 16,      kr0 + hi * 16 + 8);
      kf[1] = ld_pair(kr0 + 32 + hi * 16, kr0 + 32 + hi * 16 + 8);
      kf[2] = ld_pair(kr1 + hi * 16,      kr1 + hi * 16 + 8);
      kf[3] = ld_pair(kr1 + 32 + hi * 16, kr1 + 32 + hi * 16 + 8);
    }
    v8f s0 = {}, s1 = {};
    s0 = WMMA_F16(qa0, kf[0], s0);
    s1 = WMMA_F16(qa0, kf[2], s1);
    s0 = WMMA_F16(qa1, kf[1], s0);
    s1 = WMMA_F16(qa1, kf[3], s1);

    // online softmax: row stats via 16-lane shuffle reductions (wave32)
    float alpha[8];
#pragma unroll
    for (int r = 0; r < 8; ++r) {
      float tmax = fmaxf(s0[r], s1[r]);
#pragma unroll
      for (int msk = 1; msk < 16; msk <<= 1)
        tmax = fmaxf(tmax, __shfl_xor(tmax, msk, 32));
      const float mnew = fmaxf(mrow[r], tmax);
      alpha[r] = __expf(mrow[r] - mnew);
      mrow[r] = mnew;
      const float p0 = __expf(s0[r] - mnew);
      const float p1 = __expf(s1[r] - mnew);
      float ps = p0 + p1;
#pragma unroll
      for (int msk = 1; msk < 16; msk <<= 1)
        ps += __shfl_xor(ps, msk, 32);
      lrow[r] = lrow[r] * alpha[r] + ps;
      ldsP[w][r + hi * 8][ln]      = (h16)p0;     // P row-major for A frag
      ldsP[w][r + hi * 8][16 + ln] = (h16)p1;
    }
#pragma unroll
    for (int c = 0; c < 4; ++c)
#pragma unroll
      for (int r = 0; r < 8; ++r) o[c][r] *= alpha[r];

    // P fragment (same-wave LDS, DS ops in-order within a wave)
    v16h pf = ld_pair(&ldsP[w][ln][hi * 8], &ldsP[w][ln][16 + hi * 8]);
    __syncthreads();                              // Vt staged by all waves

    v16h vf[4];
#pragma unroll
    for (int c = 0; c < 4; ++c) {
      const h16* vp = &ldsVt[c * 16 + ln][hi * 8];
      vf[c] = ld_pair(vp, vp + 16);
    }
#pragma unroll
    for (int c = 0; c < 4; ++c) o[c] = WMMA_F16(pf, vf[c], o[c]);
  }

  // normalize and write O as f16 [b][n][h*64+d] (A-matrix for proj GEMM)
#pragma unroll
  for (int c = 0; c < 4; ++c)
#pragma unroll
    for (int r = 0; r < 8; ++r) {
      const int row = q0 + r + hi * 8;
      Ows[(size_t)(b * 1024 + row) * 768 + hh * 64 + c * 16 + ln] =
          (h16)(o[c][r] / lrow[r]);
    }
}

// ---------------------------------------------------------------------------
// Kernel 3: out = O @ W_proj + b_proj. Grid (512, 3), 128 threads.
// Block: 32 rows x 256 cols; each wave 32x64, 8 WMMA per K-step.
// ---------------------------------------------------------------------------
__global__ __launch_bounds__(128) void proj_kernel(
    const h16* __restrict__ Ows, const float* __restrict__ Wp,
    const float* __restrict__ bias, float* __restrict__ out) {
  __shared__ __align__(16) h16 ldsBt[256][40];
  const int tid  = threadIdx.x;
  const int lane = tid & 31, w = tid >> 5;
  const int ln = lane & 15, hi = lane >> 4;
  const int r0  = blockIdx.x * 32;
  const int n0b = blockIdx.y * 256;
  const int n0  = n0b + w * 64;

  v8f acc[2][4] = {};
  const h16* arow0 = Ows + (size_t)(r0 + ln) * 768;       // already f16
  const h16* arow1 = Ows + (size_t)(r0 + 16 + ln) * 768;
  for (int k0 = 0; k0 < 768; k0 += 32) {
    __syncthreads();
    for (int i = 0; i < 16; ++i) {                // stage W^T tile, float4
      int idx4 = (tid + i * 128) * 4;
      int kk = idx4 >> 8, c = idx4 & 255;
      v4f wv = *(const v4f*)&Wp[(size_t)(k0 + kk) * 768 + n0b + c];
#pragma unroll
      for (int j = 0; j < 4; ++j) ldsBt[c + j][kk] = (h16)wv[j];
    }
    if (k0 + 32 < 768)
      __builtin_prefetch(&Wp[(size_t)(k0 + 32) * 768 + n0b + tid * 4], 0, 1);
    __syncthreads();

    v16h a0 = ld_pair(arow0 + k0 + hi * 8, arow0 + k0 + hi * 8 + 16);
    v16h a1 = ld_pair(arow1 + k0 + hi * 8, arow1 + k0 + hi * 8 + 16);
    v16h bf[4];
#pragma unroll
    for (int t = 0; t < 4; ++t) {
      const h16* bp = &ldsBt[w * 64 + t * 16 + ln][hi * 8];
      bf[t] = ld_pair(bp, bp + 16);
    }
#pragma unroll
    for (int t = 0; t < 4; ++t) acc[0][t] = WMMA_F16(a0, bf[t], acc[0][t]);
#pragma unroll
    for (int t = 0; t < 4; ++t) acc[1][t] = WMMA_F16(a1, bf[t], acc[1][t]);
  }
#pragma unroll
  for (int u = 0; u < 2; ++u)
#pragma unroll
    for (int t = 0; t < 4; ++t) {
      const int col = n0 + t * 16 + ln;
      const float bv = bias[col];
#pragma unroll
      for (int r = 0; r < 8; ++r) {
        const int row = r0 + u * 16 + r + hi * 8;
        out[(size_t)row * 768 + col] = acc[u][t][r] + bv;
      }
    }
}

// ---------------------------------------------------------------------------
extern "C" void kernel_launch(void* const* d_in, const int* in_sizes, int n_in,
                              void* d_out, int out_size, void* d_ws, size_t ws_size,
                              hipStream_t stream) {
  const float* x    = (const float*)d_in[0];
  const float* sinp = (const float*)d_in[1];
  const float* cosp = (const float*)d_in[2];
  const float* Wqkv = (const float*)d_in[3];
  const float* Wp   = (const float*)d_in[4];
  const float* bp   = (const float*)d_in[5];
  float* out = (float*)d_out;

  const size_t per = (size_t)16 * 12 * 1024 * 64;  // elements per Q/K/V/O array
  h16* Qws = (h16*)d_ws;
  h16* Kws = Qws + per;
  h16* Vws = Kws + per;
  h16* Ows = Vws + per;                            // 16384 x 768 f16

  qkv_rope_kernel<<<dim3(512, 9), 128, 0, stream>>>(x, sinp, cosp, Wqkv,
                                                    Qws, Kws, Vws);
  attn_kernel<<<dim3(16, 12, 16), 128, 0, stream>>>(Qws, Kws, Vws, Ows);
  proj_kernel<<<dim3(512, 3), 128, 0, stream>>>(Ows, Wp, bp, out);
}
